// EResidualBlockBucket_73057393705089
// MI455X (gfx1250) — matrix operand order repaired and checked by reading, not compiled, not run on hardware
//
#include <hip/hip_runtime.h>
#include <hip/hip_bf16.h>

// ---------------------------------------------------------------------------
// Problem constants (from reference)
// ---------------------------------------------------------------------------
#define CCH   64          // channels
#define HW    4096        // 64*64 pixels
#define IMG   64          // H = W = 64
#define CKK   576         // C * K * K
#define ROWL  577         // CKK + bias
#define NT    216         // bucket types
#define KC    18          // CKK / 32 k-chunks
#define MAX_TILES 512     // >= 216 + 4096/16 = 472

typedef __attribute__((ext_vector_type(16))) _Float16 v16h;
typedef __attribute__((ext_vector_type(8)))  float    v8f;

// ---------------------------------------------------------------------------
// 0) zero the small counters
// ---------------------------------------------------------------------------
__global__ void zero_kernel(int* counts, int* cursor, int* nTiles) {
    int t = threadIdx.x;
    if (t < NT) { counts[t] = 0; cursor[t] = 0; }
    if (t == 0) nTiles[0] = 0;
}

// ---------------------------------------------------------------------------
// 1) body1: grouped 3x3 conv (in/out 64, groups=4 -> 16 in-ch per group) + ReLU
//    h stored channel-major: h[c*4096 + pix]
// ---------------------------------------------------------------------------
__global__ void body1_kernel(const float* __restrict__ xin,
                             const float* __restrict__ w1,
                             const float* __restrict__ b1,
                             float* __restrict__ h) {
    int idx = blockIdx.x * blockDim.x + threadIdx.x;   // c*4096 + pix
    if (idx >= CCH * HW) return;
    int c   = idx >> 12;
    int pix = idx & (HW - 1);
    int y = pix >> 6, x = pix & 63;
    int g = c >> 4;                                    // group id
    const float* wr = w1 + c * 16 * 9;
    float acc = b1[c];
#pragma unroll
    for (int ci = 0; ci < 16; ++ci) {
        const float* xc = xin + (g * 16 + ci) * HW;
#pragma unroll
        for (int t = 0; t < 9; ++t) {
            int dy = t / 3 - 1, dx = t % 3 - 1;
            int yy = y + dy, xx = x + dx;
            if ((unsigned)yy < IMG && (unsigned)xx < IMG)
                acc = fmaf(wr[ci * 9 + t], xc[yy * IMG + xx], acc);
        }
    }
    h[idx] = fmaxf(acc, 0.0f);
}

// ---------------------------------------------------------------------------
// 2) pre-pack emb filters into WMMA A-fragment order as f16 (once per launch).
//    Fragment id t = ((bucket*4 + rowtile)*KC + kchunk)*32 + lane
//    Each fragment = 16 halves = one lane's v16h A operand.
// ---------------------------------------------------------------------------
__global__ void pack_emb_kernel(const float* __restrict__ emb,
                                _Float16* __restrict__ packA) {
    int t = blockIdx.x * blockDim.x + threadIdx.x;
    if (t >= NT * 4 * KC * 32) return;
    int lane = t & 31;
    int kc   = (t >> 5) % KC;
    int rtb  = t / (KC * 32);          // bucket*4 + rowtile
    int rt   = rtb & 3;
    int b    = rtb >> 2;
    int m = lane & 15, hi = lane >> 4;
    const float* Fr = emb + (size_t)b * (CCH * ROWL) + (size_t)(rt * 16 + m) * ROWL;
    v16h frag;
#pragma unroll
    for (int i = 0; i < 16; ++i) {
        int k = kc * 32 + hi * 16 + i;   // A 16x32 f16, contiguous k order (B matches)
        frag[i] = (_Float16)Fr[k];
    }
    *(v16h*)(packA + (size_t)t * 16) = frag;
}

// ---------------------------------------------------------------------------
// 3) pre-pack w2 (64x64) into A-fragment order (4 rowtiles x 2 kchunks x 32 lanes)
// ---------------------------------------------------------------------------
__global__ void pack_w2_kernel(const float* __restrict__ w2,
                               _Float16* __restrict__ packW) {
    int t = threadIdx.x;               // 256 threads, 1 block
    int lane = t & 31;
    int kc   = (t >> 5) & 1;
    int rt   = t >> 6;
    int m = lane & 15, hi = lane >> 4;
    const float* Wr = w2 + (size_t)(rt * 16 + m) * CCH;
    v16h frag;
#pragma unroll
    for (int i = 0; i < 16; ++i) {
        int k = kc * 32 + hi * 16 + i;
        frag[i] = (_Float16)Wr[k];
    }
    *(v16h*)(packW + (size_t)t * 16) = frag;
}

// ---------------------------------------------------------------------------
// 4) bucket histogram
// ---------------------------------------------------------------------------
__global__ void count_kernel(const int* __restrict__ buckets, int* counts) {
    int pix = blockIdx.x * blockDim.x + threadIdx.x;
    if (pix < HW) atomicAdd(&counts[buckets[pix]], 1);
}

// ---------------------------------------------------------------------------
// 5) exclusive scan over 216 buckets + build 16-pixel tile descriptors
// ---------------------------------------------------------------------------
__global__ void scan_kernel(const int* __restrict__ counts,
                            int* __restrict__ offsets,
                            int* __restrict__ tiles,
                            int* __restrict__ nTiles) {
    if (threadIdx.x != 0 || blockIdx.x != 0) return;
    int run = 0, nt = 0;
    for (int b = 0; b < NT; ++b) {
        offsets[b] = run;
        int c = counts[b];
        for (int t0 = 0; t0 < c; t0 += 16)
            tiles[nt++] = (b << 16) | t0;
        run += c;
    }
    nTiles[0] = nt;
}

// ---------------------------------------------------------------------------
// 6) scatter pixel ids into per-bucket contiguous lists
// ---------------------------------------------------------------------------
__global__ void scatter_kernel(const int* __restrict__ buckets,
                               const int* __restrict__ offsets,
                               int* __restrict__ cursor,
                               int* __restrict__ pixList) {
    int pix = blockIdx.x * blockDim.x + threadIdx.x;
    if (pix >= HW) return;
    int b   = buckets[pix];
    int pos = atomicAdd(&cursor[b], 1);
    pixList[offsets[b] + pos] = pix;
}

// ---------------------------------------------------------------------------
// 7) dynamic per-pixel conv as per-bucket GEMM (WMMA hot loop):
//    O[64, 16] = F_b[64,576] x P[576, 16] + bias, mid = relu(O) stored f16.
//    A: packed fragments from global (1x 32B/lane), B: fragment-ordered LDS.
// ---------------------------------------------------------------------------
__global__ void __launch_bounds__(128)
dynconv_kernel(const float*    __restrict__ h,
               const _Float16* __restrict__ packA,
               const float*    __restrict__ emb,     // for f32 bias column
               const int*      __restrict__ tiles,
               const int*      __restrict__ nTiles,
               const int*      __restrict__ offsets,
               const int*      __restrict__ counts,
               const int*      __restrict__ pixList,
               _Float16*       __restrict__ mid) {
    __shared__ v16h PlFrag[KC * 32];   // B fragments: [kchunk][lane], 18 KB
    __shared__ int  spix[16];

    if ((int)blockIdx.x >= nTiles[0]) return;

    int desc = tiles[blockIdx.x];
    int b    = desc >> 16;
    int t0   = desc & 0xFFFF;
    int base = offsets[b];
    int cnt  = counts[b];
    int nval = min(16, cnt - t0);

    int tid = threadIdx.x;
    if (tid < 16) spix[tid] = pixList[base + t0 + min(tid, nval - 1)];
    __syncthreads();

    // stage B fragments: fragment f = kc*32 + lane; lane half holds K = kc*32+hi*16+i
    for (int f = tid; f < KC * 32; f += 128) {
        int lane = f & 31;
        int kc   = f >> 5;
        int m = lane & 15, hi = lane >> 4;
        int pix = spix[m];
        int y = pix >> 6, x = pix & 63;
        v16h frag;
#pragma unroll
        for (int i = 0; i < 16; ++i) {
            int k   = kc * 32 + hi * 16 + i;      // row of P (= c*9 + tap)
            int cch = k / 9, tap = k % 9;
            int yy = y + tap / 3 - 1, xx = x + tap % 3 - 1;
            float v = ((unsigned)yy < IMG && (unsigned)xx < IMG)
                          ? h[cch * HW + yy * IMG + xx] : 0.0f;
            frag[i] = (_Float16)v;
        }
        PlFrag[f] = frag;
    }
    __syncthreads();

    int wave = tid >> 5, lane = tid & 31;
    int m = lane & 15, hi = lane >> 4;
    int rowbase = wave * 16;

    // accumulator = per-pixel bias f[..., -1] (f32, matching reference)
    const float* F = emb + (size_t)b * (CCH * ROWL);
    v8f acc;
#pragma unroll
    for (int r = 0; r < 8; ++r)
        acc[r] = F[(rowbase + r + hi * 8) * ROWL + CKK];

    // hot loop: 18x { 32B A load, 2x ds_load_b128, wmma }
    const v16h* Af = (const v16h*)(packA) + ((size_t)(b * 4 + wave) * KC) * 32 + lane;
    const v16h* Bf = PlFrag + lane;
    for (int kc = 0; kc < KC; ++kc) {
        v16h a  = Af[kc * 32];
        v16h bm = Bf[kc * 32];
        acc = __builtin_amdgcn_wmma_f32_16x16x32_f16(
            false, a, false, bm, (short)0, acc, false, false);
    }

    // mid = relu(O) stored as f16 (only ever consumed as a f16 WMMA B operand)
#pragma unroll
    for (int r = 0; r < 8; ++r) {
        int row = rowbase + r + hi * 8;
        if (m < nval) mid[row * HW + spix[m]] = (_Float16)fmaxf(acc[r], 0.0f);
    }
}

// ---------------------------------------------------------------------------
// 8) body2: out = relu( W2[64,64] x mid[64,4096] + b2 + x )  -- WMMA GEMM
// ---------------------------------------------------------------------------
__global__ void __launch_bounds__(128)
conv1x1_kernel(const _Float16* __restrict__ mid,
               const _Float16* __restrict__ packW,
               const float*    __restrict__ b2,
               const float*    __restrict__ xin,
               float* __restrict__ out) {
    int pixbase = blockIdx.x * 16;
    int tid = threadIdx.x;
    int wave = tid >> 5, lane = tid & 31;
    int m = lane & 15, hi = lane >> 4;
    int rowbase = wave * 16;

    v8f acc;
#pragma unroll
    for (int r = 0; r < 8; ++r)
        acc[r] = b2[rowbase + r + hi * 8];

    const v16h* Wf = (const v16h*)(packW) + (size_t)wave * 2 * 32 + lane;
#pragma unroll
    for (int kci = 0; kci < 2; ++kci) {
        v16h a = Wf[kci * 32];
        v16h bm;
#pragma unroll
        for (int i = 0; i < 16; ++i) {
            int k = kci * 32 + hi * 16 + i;
            bm[i] = mid[k * HW + pixbase + m];     // f16, coalesced across lanes
        }
        acc = __builtin_amdgcn_wmma_f32_16x16x32_f16(
            false, a, false, bm, (short)0, acc, false, false);
    }

#pragma unroll
    for (int r = 0; r < 8; ++r) {
        int row = rowbase + r + hi * 8;
        int p   = row * HW + pixbase + m;
        out[p] = fmaxf(acc[r] + xin[p], 0.0f);
    }
}

// ---------------------------------------------------------------------------
// host launcher
// ---------------------------------------------------------------------------
static inline char* carve(char*& p, size_t bytes) {
    char* r = p;
    p += (bytes + 255) & ~size_t(255);
    return r;
}

extern "C" void kernel_launch(void* const* d_in, const int* in_sizes, int n_in,
                              void* d_out, int out_size, void* d_ws, size_t ws_size,
                              hipStream_t stream) {
    const float* x       = (const float*)d_in[0];
    const int*   buckets = (const int*)  d_in[1];
    const float* w1      = (const float*)d_in[2];
    const float* b1      = (const float*)d_in[3];
    const float* emb     = (const float*)d_in[4];
    const float* w2      = (const float*)d_in[5];
    const float* b2      = (const float*)d_in[6];
    float* out = (float*)d_out;

    // workspace carve (~17.5 MB)
    char* p = (char*)d_ws;
    float*     h       = (float*)    carve(p, sizeof(float) * CCH * HW);
    _Float16*  mid     = (_Float16*) carve(p, sizeof(_Float16) * CCH * HW);
    _Float16*  packA   = (_Float16*) carve(p, sizeof(_Float16) * (size_t)NT * 4 * KC * 32 * 16);
    _Float16*  packW   = (_Float16*) carve(p, sizeof(_Float16) * 4 * 2 * 32 * 16);
    int*       counts  = (int*)      carve(p, sizeof(int) * NT);
    int*       offsets = (int*)      carve(p, sizeof(int) * NT);
    int*       cursor  = (int*)      carve(p, sizeof(int) * NT);
    int*       pixList = (int*)      carve(p, sizeof(int) * HW);
    int*       tiles   = (int*)      carve(p, sizeof(int) * MAX_TILES);
    int*       nTiles  = (int*)      carve(p, sizeof(int) * 1);
    (void)ws_size; (void)in_sizes; (void)n_in; (void)out_size;

    zero_kernel<<<1, 256, 0, stream>>>(counts, cursor, nTiles);
    body1_kernel<<<(CCH * HW) / 256, 256, 0, stream>>>(x, w1, b1, h);
    pack_emb_kernel<<<(NT * 4 * KC * 32 + 255) / 256, 256, 0, stream>>>(emb, packA);
    pack_w2_kernel<<<1, 256, 0, stream>>>(w2, packW);
    count_kernel<<<HW / 256, 256, 0, stream>>>(buckets, counts);
    scan_kernel<<<1, 32, 0, stream>>>(counts, offsets, tiles, nTiles);
    scatter_kernel<<<HW / 256, 256, 0, stream>>>(buckets, offsets, cursor, pixList);
    dynconv_kernel<<<MAX_TILES, 128, 0, stream>>>(h, packA, emb, tiles, nTiles,
                                                  offsets, counts, pixList, mid);
    conv1x1_kernel<<<HW / 16, 128, 0, stream>>>(mid, packW, b2, x, out);
}
